// LATTE_9414568313294
// MI455X (gfx1250) — matrix operand order repaired and checked by reading, not compiled, not run on hardware
//
#include <hip/hip_runtime.h>
#include <hip/hip_bf16.h>

typedef __attribute__((ext_vector_type(16))) __bf16 v16bf;
typedef __attribute__((ext_vector_type(8)))  float  v8f;

// ---------------------------------------------------------------------------
// Zero-fill (grid-stride)
// ---------------------------------------------------------------------------
__global__ void zero_f32(float* __restrict__ p, long n) {
    long i = (long)blockIdx.x * blockDim.x + threadIdx.x;
    long stride = (long)gridDim.x * blockDim.x;
    for (; i < n; i += stride) p[i] = 0.0f;
}

// ---------------------------------------------------------------------------
// Degree histogram over the shared dimension (Nb)
// ---------------------------------------------------------------------------
__global__ void degree_kernel(const long long* __restrict__ eab,
                              const long long* __restrict__ eba,
                              float* __restrict__ deg, int E) {
    int i = blockIdx.x * blockDim.x + threadIdx.x;
    if (i < E) {
        atomicAdd(&deg[(int)eab[(long)E + i]], 1.0f);  // edge_index_ab[1][i]
        atomicAdd(&deg[(int)eba[i]], 1.0f);            // edge_index_ba[0][i]
    }
}

__global__ void recip_kernel(const float* __restrict__ deg,
                             float* __restrict__ dvec, int n) {
    int i = blockIdx.x * blockDim.x + threadIdx.x;
    if (i < n) {
        float d = deg[i];
        dvec[i] = (d > 0.0f) ? (1.0f / d) : 0.0f;
    }
}

// ---------------------------------------------------------------------------
// Pre-pack W [K,N] f32 -> bf16 fragments in wave32 B-fragment order.
// Tile (kt, nt) covers K rows [kt*32, kt*32+32), N cols [nt*16, nt*16+16).
// Fragment map (ISA 7.12.2): lane l -> col nt*16 + l%16; elem e -> K = kt*32 + (e<8?0:16) + 8*(l/16) + (e&7)
// Packed index: (((kt*NT + nt)*32 + lane)*16 + e), so each lane reads one
// contiguous, 32B-aligned v16bf in the GEMM.
// ---------------------------------------------------------------------------
__global__ void pack_W_bf16(const float* __restrict__ W, __bf16* __restrict__ P,
                            int K, int N) {
    int idx = blockIdx.x * blockDim.x + threadIdx.x;  // one thread per element
    if (idx >= K * N) return;
    int e    = idx & 15;
    int lane = (idx >> 4) & 31;
    int tile = idx >> 9;
    int NT   = N >> 4;
    int nt   = tile % NT;
    int kt   = tile / NT;
    int kk   = ((e & 8) << 1) + ((lane >> 4) << 3) + (e & 7);  // (e<8?0:16) + 8*(lane/16) + (e&7)
    P[idx] = (__bf16)W[(long)(kt * 32 + kk) * N + nt * 16 + (lane & 15)];
}

// ---------------------------------------------------------------------------
// Projection GEMM + bias + relu via v_wmma_f32_16x16x32_bf16 (wave32).
// H[M,N] = relu(X[M,K] @ W[K,N] + b[N]),  K % 32 == 0, N == 128.
// Block = 128 threads (4 waves), 2 M-tiles per block (32 rows).
// Each wave owns two 16-wide N tiles -> 4 WMMAs per k-step, B fragments
// loaded once (vector v16bf loads from the pre-packed array).
//
// A fragment (16x32 bf16): lane l -> row l%16;
//   elem e -> K = (e<8?0:16) + 8*(l/16) + (e&7): two contiguous 8-float runs,
//   fetched as 4 aligned float4 loads then converted.
// C/D (16x16 f32): vgpr i, lane l -> M = i + 8*(l/16), N = l%16
// ---------------------------------------------------------------------------
__device__ inline void cvt8_to(v16bf& a, int base, float4 x, float4 y) {
    a[base + 0] = (__bf16)x.x; a[base + 1] = (__bf16)x.y;
    a[base + 2] = (__bf16)x.z; a[base + 3] = (__bf16)x.w;
    a[base + 4] = (__bf16)y.x; a[base + 5] = (__bf16)y.y;
    a[base + 6] = (__bf16)y.z; a[base + 7] = (__bf16)y.w;
}

__global__ __launch_bounds__(128)
void proj_gemm_relu(const float* __restrict__ X, const __bf16* __restrict__ Wp,
                    const float* __restrict__ bias, float* __restrict__ H,
                    int K, int N) {
    const int wave = threadIdx.x >> 5;
    const int lane = threadIdx.x & 31;
    const int row  = lane & 15;
    const int half = lane >> 4;
    const int m0   = blockIdx.x * 32;      // 2 M-tiles per block
    const int nt0  = wave * 2;             // 2 N-tiles per wave
    const int NT   = N >> 4;

    v8f acc00 = {}, acc01 = {}, acc10 = {}, acc11 = {};

    const float* xr0 = X + (long)(m0 + row) * K;
    const float* xr1 = xr0 + (long)16 * K;
    const int KT = K >> 5;

    for (int kt = 0; kt < KT; ++kt) {
        const int k0 = kt * 32;
        const int kb = k0 + half * 8;      // 32B-aligned base for this lane

        // ---- A fragments: 4x float4 vector loads each, cvt to bf16 ----
        float4 a0c0 = *(const float4*)(xr0 + kb);
        float4 a0c1 = *(const float4*)(xr0 + kb + 4);
        float4 a0c2 = *(const float4*)(xr0 + kb + 16);
        float4 a0c3 = *(const float4*)(xr0 + kb + 20);
        float4 a1c0 = *(const float4*)(xr1 + kb);
        float4 a1c1 = *(const float4*)(xr1 + kb + 4);
        float4 a1c2 = *(const float4*)(xr1 + kb + 16);
        float4 a1c3 = *(const float4*)(xr1 + kb + 20);

        // ---- B fragments: single contiguous 32B vector load each ----
        const v16bf b0 = *(const v16bf*)(Wp + ((long)(kt * NT + nt0)     * 32 + lane) * 16);
        const v16bf b1 = *(const v16bf*)(Wp + ((long)(kt * NT + nt0 + 1) * 32 + lane) * 16);

        v16bf a0, a1;
        cvt8_to(a0, 0, a0c0, a0c1); cvt8_to(a0, 8, a0c2, a0c3);
        cvt8_to(a1, 0, a1c0, a1c1); cvt8_to(a1, 8, a1c2, a1c3);

        acc00 = __builtin_amdgcn_wmma_f32_16x16x32_bf16(false, a0, false, b0, (short)0, acc00, false, false);
        acc01 = __builtin_amdgcn_wmma_f32_16x16x32_bf16(false, a0, false, b1, (short)0, acc01, false, false);
        acc10 = __builtin_amdgcn_wmma_f32_16x16x32_bf16(false, a1, false, b0, (short)0, acc10, false, false);
        acc11 = __builtin_amdgcn_wmma_f32_16x16x32_bf16(false, a1, false, b1, (short)0, acc11, false, false);
    }

    const int c0 = nt0 * 16 + row;
    const int c1 = c0 + 16;
    const float bias0 = bias[c0];
    const float bias1 = bias[c1];
#pragma unroll
    for (int i = 0; i < 8; ++i) {
        int mA = m0 + i + half * 8;
        int mB = mA + 16;
        float v;
        v = acc00[i] + bias0; H[(long)mA * N + c0] = v > 0.0f ? v : 0.0f;
        v = acc01[i] + bias1; H[(long)mA * N + c1] = v > 0.0f ? v : 0.0f;
        v = acc10[i] + bias0; H[(long)mB * N + c0] = v > 0.0f ? v : 0.0f;
        v = acc11[i] + bias1; H[(long)mB * N + c1] = v > 0.0f ? v : 0.0f;
    }
}

// ---------------------------------------------------------------------------
// t = B_mat @ h_a : for each edge_ba (r, c): t[r, :] += h_a[c, :]
// ---------------------------------------------------------------------------
__global__ void spmm_scatter_ba(const long long* __restrict__ eba,
                                const float* __restrict__ h_a,
                                float* __restrict__ t, int E, int D) {
    long gid = (long)blockIdx.x * blockDim.x + threadIdx.x;
    if (gid >= (long)E * D) return;
    int e = (int)(gid / D);
    int d = (int)(gid % D);
    int r = (int)eba[e];
    int c = (int)eba[(long)E + e];
    atomicAdd(&t[(long)r * D + d], h_a[(long)c * D + d]);
}

// ---------------------------------------------------------------------------
// out = (A*d) @ t + A @ h_b : for each edge_ab (i, j):
//   out[i, :] += dvec[j] * t[j, :] + h_b[j, :]
// ---------------------------------------------------------------------------
__global__ void final_scatter_ab(const long long* __restrict__ eab,
                                 const float* __restrict__ t,
                                 const float* __restrict__ h_b,
                                 const float* __restrict__ dvec,
                                 float* __restrict__ out, int E, int D) {
    long gid = (long)blockIdx.x * blockDim.x + threadIdx.x;
    if (gid >= (long)E * D) return;
    int e = (int)(gid / D);
    int d = (int)(gid % D);
    int i = (int)eab[e];
    int j = (int)eab[(long)E + e];
    float contrib = dvec[j] * t[(long)j * D + d] + h_b[(long)j * D + d];
    atomicAdd(&out[(long)i * D + d], contrib);
}

// ---------------------------------------------------------------------------
// Host launcher
// ---------------------------------------------------------------------------
extern "C" void kernel_launch(void* const* d_in, const int* in_sizes, int n_in,
                              void* d_out, int out_size, void* d_ws, size_t ws_size,
                              hipStream_t stream) {
    const float*     x_a = (const float*)d_in[0];
    const float*     x_b = (const float*)d_in[1];
    const float*     W_a = (const float*)d_in[2];
    const float*     b_a = (const float*)d_in[3];
    const float*     W_b = (const float*)d_in[4];
    const float*     b_b = (const float*)d_in[5];
    const long long* eab = (const long long*)d_in[6];
    const long long* eba = (const long long*)d_in[7];

    const int D  = in_sizes[3];            // 128
    const int Fa = in_sizes[2] / D;        // 512
    const int Fb = in_sizes[4] / D;        // 512
    const int Na = in_sizes[0] / Fa;       // 4096
    const int Nb = in_sizes[1] / Fb;       // 4096
    const int E  = in_sizes[6] / 2;        // 131072

    // Workspace layout (floats): h_a | h_b | t | deg | dvec | Wp_a | Wp_b (bf16)
    float* ws    = (float*)d_ws;
    float* h_a   = ws;
    float* h_b   = h_a + (size_t)Na * D;
    float* t     = h_b + (size_t)Nb * D;
    float* deg   = t   + (size_t)Nb * D;
    float* dvec  = deg + Nb;
    __bf16* Wp_a = (__bf16*)(dvec + Nb);
    __bf16* Wp_b = Wp_a + (size_t)Fa * D;
    float* out   = (float*)d_out;

    // 1) zero t + deg (contiguous) and out
    {
        long nz = (long)Nb * D + Nb;
        int blocks = (int)((nz + 255) / 256); if (blocks > 4096) blocks = 4096;
        zero_f32<<<blocks, 256, 0, stream>>>(t, nz);
        long no = (long)Na * D;
        int blocks2 = (int)((no + 255) / 256); if (blocks2 > 4096) blocks2 = 4096;
        zero_f32<<<blocks2, 256, 0, stream>>>(out, no);
    }

    // 2) degree histogram + reciprocal
    degree_kernel<<<(E + 255) / 256, 256, 0, stream>>>(eab, eba, deg, E);
    recip_kernel<<<(Nb + 255) / 256, 256, 0, stream>>>(deg, dvec, Nb);

    // 3) pre-pack weights into bf16 WMMA B-fragment layout
    pack_W_bf16<<<(Fa * D + 255) / 256, 256, 0, stream>>>(W_a, Wp_a, Fa, D);
    pack_W_bf16<<<(Fb * D + 255) / 256, 256, 0, stream>>>(W_b, Wp_b, Fb, D);

    // 4) feature projections via WMMA (bf16 in, f32 accumulate)
    proj_gemm_relu<<<Na / 32, 128, 0, stream>>>(x_a, Wp_a, b_a, h_a, Fa, D);
    proj_gemm_relu<<<Nb / 32, 128, 0, stream>>>(x_b, Wp_b, b_b, h_b, Fb, D);

    // 5) t = B @ h_a   (scatter over edge_ba)
    {
        long nthreads = (long)E * D;
        int blocks = (int)((nthreads + 255) / 256);
        spmm_scatter_ba<<<blocks, 256, 0, stream>>>(eba, h_a, t, E, D);
    }

    // 6) out = (A*d) @ t + A @ h_b   (scatter over edge_ab)
    {
        long nthreads = (long)E * D;
        int blocks = (int)((nthreads + 255) / 256);
        final_scatter_ab<<<blocks, 256, 0, stream>>>(eab, t, h_b, dvec, out, E, D);
    }
}
